// MultiHeadAttention_80504866996371
// MI455X (gfx1250) — matrix operand order repaired and checked by reading, not compiled, not run on hardware
//
#include <hip/hip_runtime.h>

#define DEVINL __device__ __forceinline__

typedef __attribute__((ext_vector_type(16))) __bf16 v16bf;
typedef __attribute__((ext_vector_type(8)))  float  v8f;

union Frag {
  v16bf v;
  unsigned short u[16];
  uint4 q[2];
};

constexpr int B_ = 4, T_ = 2048, C_ = 1024, H_ = 16, D_ = 64;
constexpr int BT_ = B_ * T_;
constexpr float SCALE_ = 0.03125f;  // C^-0.5 = 1/32 (reference scales by n_embd)

DEVINL unsigned short f2bf(float f) {
  union { float f; unsigned u; } x; x.f = f;
  unsigned r = x.u + 0x7FFFu + ((x.u >> 16) & 1u);  // round-to-nearest-even
  return (unsigned short)(r >> 16);
}

DEVINL v8f vzero() {
  v8f z;
#pragma unroll
  for (int i = 0; i < 8; ++i) z[i] = 0.f;
  return z;
}

// A-matrix 16x32 bf16 fragment. base -> (row0, k0), row-major, stride in halfs.
// ISA layout: lane<16: row=lane, elems 0..7 => K 0..7, elems 8..15 => K 16..23
//             lane>=16: row=lane-16,      => K 8..15,            => K 24..31
DEVINL Frag load_a16(const unsigned short* base, int stride, int lane) {
  Frag f;
  int r = lane & 15;
  int o = (lane >> 4) ? 8 : 0;
  f.q[0] = *(const uint4*)(base + r * stride + o);
  f.q[1] = *(const uint4*)(base + r * stride + o + 16);
  return f;
}

// B-matrix 32x16 bf16 fragment from K-contiguous storage: base -> (col0, k0),
// row index = output column n, stride in halfs.
// ISA layout: lane<16: col=lane, elems 0..7 => K 0..7, 8..15 => K 16..23 ; lane>=16: col=lane-16 => K 8..15, K 24..31
DEVINL Frag load_b16(const unsigned short* base, int stride, int lane) {
  Frag f;
  int c = lane & 15;
  int o = (lane >> 4) ? 8 : 0;
  f.q[0] = *(const uint4*)(base + c * stride + o);
  f.q[1] = *(const uint4*)(base + c * stride + o + 16);
  return f;
}

DEVINL v8f wmma_bf16(const Frag& a, const Frag& b, v8f c) {
  return __builtin_amdgcn_wmma_f32_16x16x32_bf16(false, a.v, false, b.v,
                                                 (short)0, c, false, false);
}

// ---------------- conversion kernels ----------------

__global__ void cvt_bf16_kernel(const float* __restrict__ s,
                                unsigned short* __restrict__ d, int n) {
  int i = blockIdx.x * blockDim.x + threadIdx.x;
  if (i < n) d[i] = f2bf(s[i]);
}

// [H][C][D] f32 -> [H][D][C] bf16 (per-head transpose for K-contiguous B frags)
__global__ void cvt_headsT_kernel(const float* __restrict__ s,
                                  unsigned short* __restrict__ d) {
  int i = blockIdx.x * blockDim.x + threadIdx.x;  // i < H*C*D
  int h = i >> 16;               // C*D = 65536
  int rem = i & 65535;
  int c = rem >> 6, dd = rem & 63;
  d[(size_t)((h << 6) + dd) * C_ + c] = f2bf(s[i]);
}

// [C][C] f32 -> transposed [C_out][C_in] bf16
__global__ void cvt_wpT_kernel(const float* __restrict__ s,
                               unsigned short* __restrict__ d) {
  int i = blockIdx.x * blockDim.x + threadIdx.x;  // i < C*C
  int r = i >> 10, c = i & 1023;
  d[(size_t)c * C_ + r] = f2bf(s[i]);
}

// ---------------- QKV projection: q/k/v[b,h,t,d] = x[b,t,:] @ W*[h] ----------

__global__ void __launch_bounds__(256) qkv_gemm_kernel(
    const unsigned short* __restrict__ xb,
    const unsigned short* __restrict__ wqt,
    const unsigned short* __restrict__ wkt,
    const unsigned short* __restrict__ wvt,
    unsigned short* __restrict__ qw,
    unsigned short* __restrict__ kw,
    unsigned short* __restrict__ vw) {
  __shared__ __align__(16) unsigned short xs[128][64];      // 16 KB
  __shared__ __align__(16) unsigned short wls[3][64][64];   // 24 KB
  const int tid = threadIdx.x;
  const int lane = tid & 31, w = tid >> 5;
  const int ln = lane & 15, hs = lane >> 4;
  const int h = blockIdx.y;
  const int rowbase = blockIdx.x * 128;

  v8f acc[3][4];
#pragma unroll
  for (int m = 0; m < 3; ++m)
#pragma unroll
    for (int n = 0; n < 4; ++n) acc[m][n] = vzero();

  for (int kc = 0; kc < C_; kc += 64) {  // 16 stages, 24 WMMAs per stage
    __syncthreads();
#pragma unroll
    for (int it = 0; it < 4; ++it) {           // 1024 chunks of 8 halfs
      int idx = tid + it * 256;
      int row = idx >> 3, part = idx & 7;
      *(uint4*)&xs[row][part * 8] =
          *(const uint4*)(xb + (size_t)(rowbase + row) * C_ + kc + part * 8);
    }
#pragma unroll
    for (int it = 0; it < 6; ++it) {           // 3 * 64 * 8 chunks
      int idx = tid + it * 256;
      int mat = idx >> 9, rem = idx & 511;
      int row = rem >> 3, part = rem & 7;
      const unsigned short* src = (mat == 0) ? wqt : ((mat == 1) ? wkt : wvt);
      *(uint4*)&wls[mat][row][part * 8] =
          *(const uint4*)(src + (size_t)(h * D_ + row) * C_ + kc + part * 8);
    }
    __syncthreads();
    Frag a[2];
#pragma unroll
    for (int kk = 0; kk < 2; ++kk)
      a[kk] = load_a16(&xs[w * 16][kk * 32], 64, lane);
#pragma unroll
    for (int m = 0; m < 3; ++m)
#pragma unroll
      for (int n = 0; n < 4; ++n)
#pragma unroll
        for (int kk = 0; kk < 2; ++kk) {
          Frag b = load_b16(&wls[m][n * 16][kk * 32], 64, lane);
          acc[m][n] = wmma_bf16(a[kk], b, acc[m][n]);
        }
  }
  unsigned short* outp[3] = {qw, kw, vw};
#pragma unroll
  for (int m = 0; m < 3; ++m)
#pragma unroll
    for (int n = 0; n < 4; ++n)
#pragma unroll
      for (int r = 0; r < 8; ++r) {
        int grow = rowbase + w * 16 + hs * 8 + r;
        int b = grow >> 11, t = grow & (T_ - 1);
        outp[m][(size_t)((b * H_ + h) * T_ + t) * D_ + n * 16 + ln] =
            f2bf(acc[m][n][r]);
      }
}

// ---------------- flash attention (causal, streaming softmax) ---------------

__global__ void __launch_bounds__(256) attn_kernel(
    const unsigned short* __restrict__ qw,
    const unsigned short* __restrict__ kw,
    const unsigned short* __restrict__ vw,
    unsigned short* __restrict__ ow) {
  __shared__ __align__(16) unsigned short ks[64][72];      // K tile, row-major
  __shared__ __align__(16) unsigned short vt[64][72];      // V tile, transposed
  __shared__ __align__(16) unsigned short pls[8][16][72];  // per-wave P strip
  const int tid = threadIdx.x;
  const int lane = tid & 31, w = tid >> 5;
  const int ln = lane & 15, hs = lane >> 4;
  const int h = blockIdx.y, b = blockIdx.z;
  const int qbase = blockIdx.x * 128;
  const size_t bh = (size_t)(b * H_ + h) * T_ * D_;
  const unsigned short* qg = qw + bh;
  const unsigned short* kg = kw + bh;
  const unsigned short* vg = vw + bh;

  // Preload this wave's 16 q rows as two 16x32 A fragments (reused every block)
  Frag aq[2];
  {
    const unsigned short* qrow = qg + (size_t)(qbase + w * 16 + ln) * D_;
    int o = hs ? 8 : 0;
#pragma unroll
    for (int i = 0; i < 2; ++i) {
      aq[i].q[0] = *(const uint4*)(qrow + i * 32 + o);
      aq[i].q[1] = *(const uint4*)(qrow + i * 32 + o + 16);
    }
  }

  v8f acco[4];
#pragma unroll
  for (int n = 0; n < 4; ++n) acco[n] = vzero();
  float mrow[8], lrow[8];
#pragma unroll
  for (int r = 0; r < 8; ++r) { mrow[r] = -1e30f; lrow[r] = 0.f; }

  const int nkb = (qbase >> 6) + 2;  // causal: key blocks of 64 up to diagonal
  for (int kb = 0; kb < nkb; ++kb) {
    __syncthreads();
#pragma unroll
    for (int it = 0; it < 2; ++it) {  // 512 chunks of 8 halfs each for K and V
      int idx = tid + it * 256;
      int s = idx >> 3, part = idx & 7;
      *(uint4*)&ks[s][part * 8] =
          *(const uint4*)(kg + (size_t)(kb * 64 + s) * D_ + part * 8);
      uint4 dv = *(const uint4*)(vg + (size_t)(kb * 64 + s) * D_ + part * 8);
      const unsigned short* pv = (const unsigned short*)&dv;
#pragma unroll
      for (int j = 0; j < 8; ++j) vt[part * 8 + j][s] = pv[j];
    }
    __syncthreads();

    // prefetch next key block's K/V rows (gfx1250 global_prefetch_b8 path);
    // overlaps HBM/L2 fetch with this block's WMMAs + softmax
    if (kb + 1 < nkb && tid < 128) {
      int r = tid & 63;
      const unsigned short* pf =
          ((tid < 64) ? kg : vg) + (size_t)((kb + 1) * 64 + r) * D_;
      __builtin_prefetch(pf, 0, 3);
    }

    const bool maskblk = (kb * 64 + 63) > qbase;

    // S = q @ k^T  (k row-major is already K-contiguous for the B fragment)
    v8f S[4];
#pragma unroll
    for (int n = 0; n < 4; ++n) {
      v8f s = vzero();
#pragma unroll
      for (int dk = 0; dk < 2; ++dk) {
        Frag bk = load_b16(&ks[n * 16][dk * 32], 72, lane);
        s = wmma_bf16(aq[dk], bk, s);
      }
      S[n] = s;
    }
    // scale + causal mask
#pragma unroll
    for (int n = 0; n < 4; ++n) {
      int col = kb * 64 + n * 16 + ln;
#pragma unroll
      for (int r = 0; r < 8; ++r) {
        float sv = S[n][r] * SCALE_;
        int rowt = qbase + w * 16 + hs * 8 + r;
        if (maskblk && col > rowt) sv = -1e30f;
        S[n][r] = sv;
      }
    }
    // online softmax: row max / rescale / exp / row sum
    float cf[8], rs[8];
#pragma unroll
    for (int r = 0; r < 8; ++r) {
      float x = fmaxf(fmaxf(S[0][r], S[1][r]), fmaxf(S[2][r], S[3][r]));
#pragma unroll
      for (int msk = 1; msk <= 8; msk <<= 1)
        x = fmaxf(x, __shfl_xor(x, msk, 32));
      float mn = fmaxf(mrow[r], x);
      cf[r] = __expf(mrow[r] - mn);
      mrow[r] = mn;
      rs[r] = 0.f;
    }
#pragma unroll
    for (int n = 0; n < 4; ++n)
#pragma unroll
      for (int r = 0; r < 8; ++r) {
        float p = __expf(S[n][r] - mrow[r]);
        S[n][r] = p;
        rs[r] += p;
      }
#pragma unroll
    for (int r = 0; r < 8; ++r) {
      float x = rs[r];
#pragma unroll
      for (int msk = 1; msk <= 8; msk <<= 1) x += __shfl_xor(x, msk, 32);
      lrow[r] = lrow[r] * cf[r] + x;
    }
#pragma unroll
    for (int n = 0; n < 4; ++n)
#pragma unroll
      for (int r = 0; r < 8; ++r) acco[n][r] *= cf[r];

    // P: C-layout f32 -> wave-private LDS strip as bf16 (A-layout reload)
#pragma unroll
    for (int n = 0; n < 4; ++n)
#pragma unroll
      for (int r = 0; r < 8; ++r)
        pls[w][hs * 8 + r][n * 16 + ln] = f2bf(S[n][r]);

    // acco += P @ V  (vt is transposed -> K-contiguous B fragments)
#pragma unroll
    for (int sk = 0; sk < 2; ++sk) {
      Frag ap = load_a16(&pls[w][0][sk * 32], 72, lane);
#pragma unroll
      for (int n = 0; n < 4; ++n) {
        Frag bv = load_b16(&vt[n * 16][sk * 32], 72, lane);
        acco[n] = wmma_bf16(ap, bv, acco[n]);
      }
    }
  }
  // normalize, store O at [b][t][h*64+d] (heads concatenated)
#pragma unroll
  for (int n = 0; n < 4; ++n)
#pragma unroll
    for (int r = 0; r < 8; ++r) {
      int rowt = qbase + w * 16 + hs * 8 + r;
      float val = acco[n][r] / lrow[r];
      ow[(size_t)(b * T_ + rowt) * C_ + h * D_ + n * 16 + ln] = f2bf(val);
    }
}

// ---------------- output projection: Y = O @ Wp + bp ------------------------

__global__ void __launch_bounds__(256) proj_kernel(
    const unsigned short* __restrict__ ob,
    const unsigned short* __restrict__ wpt,
    const float* __restrict__ bp,
    float* __restrict__ out) {
  __shared__ __align__(16) unsigned short os[128][64];     // 16 KB
  __shared__ __align__(16) unsigned short wls[64][64];     // 8 KB
  const int tid = threadIdx.x;
  const int lane = tid & 31, w = tid >> 5;
  const int ln = lane & 15, hs = lane >> 4;
  const int rowbase = blockIdx.x * 128;
  const int nbase = blockIdx.y * 64;

  v8f acc[4];
#pragma unroll
  for (int n = 0; n < 4; ++n) acc[n] = vzero();

  for (int kc = 0; kc < C_; kc += 64) {  // 16 stages, 8 WMMAs per stage
    __syncthreads();
#pragma unroll
    for (int it = 0; it < 4; ++it) {           // 1024 chunks
      int idx = tid + it * 256;
      int row = idx >> 3, part = idx & 7;
      *(uint4*)&os[row][part * 8] =
          *(const uint4*)(ob + (size_t)(rowbase + row) * C_ + kc + part * 8);
    }
#pragma unroll
    for (int it = 0; it < 2; ++it) {           // 512 chunks
      int idx = tid + it * 256;
      int row = idx >> 3, part = idx & 7;
      *(uint4*)&wls[row][part * 8] =
          *(const uint4*)(wpt + (size_t)(nbase + row) * C_ + kc + part * 8);
    }
    __syncthreads();
    Frag a[2];
#pragma unroll
    for (int kk = 0; kk < 2; ++kk)
      a[kk] = load_a16(&os[w * 16][kk * 32], 64, lane);
#pragma unroll
    for (int n = 0; n < 4; ++n)
#pragma unroll
      for (int kk = 0; kk < 2; ++kk) {
        Frag bfr = load_b16(&wls[n * 16][kk * 32], 64, lane);
        acc[n] = wmma_bf16(a[kk], bfr, acc[n]);
      }
  }
#pragma unroll
  for (int n = 0; n < 4; ++n) {
    int col = nbase + n * 16 + ln;
    float bias = bp[col];
#pragma unroll
    for (int r = 0; r < 8; ++r) {
      int grow = rowbase + w * 16 + hs * 8 + r;
      out[(size_t)grow * C_ + col] = acc[n][r] + bias;
    }
  }
}

// ---------------- host launch ----------------------------------------------

extern "C" void kernel_launch(void* const* d_in, const int* in_sizes, int n_in,
                              void* d_out, int out_size, void* d_ws,
                              size_t ws_size, hipStream_t stream) {
  (void)in_sizes; (void)n_in; (void)out_size; (void)ws_size;
  const float* x  = (const float*)d_in[0];
  const float* Wq = (const float*)d_in[1];
  const float* Wk = (const float*)d_in[2];
  const float* Wv = (const float*)d_in[3];
  const float* Wp = (const float*)d_in[4];
  const float* bp = (const float*)d_in[5];
  float* out = (float*)d_out;

  constexpr int NX = BT_ * C_;          // 8,388,608
  constexpr int NWH = H_ * C_ * D_;     // 1,048,576 per head-weight tensor
  constexpr int NWP = C_ * C_;          // 1,048,576
  constexpr int NQ = B_ * H_ * T_ * D_; // 8,388,608

  unsigned short* p = (unsigned short*)d_ws;   // total ~88 MB of ws
  unsigned short* xb  = p; p += NX;
  unsigned short* wqt = p; p += NWH;
  unsigned short* wkt = p; p += NWH;
  unsigned short* wvt = p; p += NWH;
  unsigned short* wpt = p; p += NWP;
  unsigned short* qw  = p; p += NQ;
  unsigned short* kw  = p; p += NQ;
  unsigned short* vw  = p; p += NQ;
  unsigned short* ow  = p; p += NX;

  cvt_bf16_kernel<<<NX / 256, 256, 0, stream>>>(x, xb, NX);
  cvt_headsT_kernel<<<NWH / 256, 256, 0, stream>>>(Wq, wqt);
  cvt_headsT_kernel<<<NWH / 256, 256, 0, stream>>>(Wk, wkt);
  cvt_headsT_kernel<<<NWH / 256, 256, 0, stream>>>(Wv, wvt);
  cvt_wpT_kernel<<<NWP / 256, 256, 0, stream>>>(Wp, wpt);

  qkv_gemm_kernel<<<dim3(BT_ / 128, H_), 256, 0, stream>>>(xb, wqt, wkt, wvt,
                                                           qw, kw, vw);
  attn_kernel<<<dim3(T_ / 128, H_, B_), 256, 0, stream>>>(qw, kw, vw, ow);
  proj_kernel<<<dim3(BT_ / 128, C_ / 64), 256, 0, stream>>>(ow, wpt, bp, out);
}
